// RelationalGraphletConvolutionGroupAttn_4372276708026
// MI455X (gfx1250) — matrix operand order repaired and checked by reading, not compiled, not run on hardware
//
#include <hip/hip_runtime.h>
#include <hip/hip_bf16.h>

// ---------------------------------------------------------------------------
// Problem constants (from reference)
// ---------------------------------------------------------------------------
#define BB   128
#define NN   1024
#define DD   256
#define GG   64
#define GSZ  3
#define QQ   (GG * GSZ)     // 192
#define REL  16
#define KDIM 64
#define PDIM 64
#define NF   64

typedef __bf16  bf16x16 __attribute__((ext_vector_type(16)));
typedef __bf16  bf16x8  __attribute__((ext_vector_type(8)));
typedef float   v8f     __attribute__((ext_vector_type(8)));

// ---------------------------------------------------------------------------
// WMMA helpers. CDNA5 wave32 layouts (CDNA5 ISA §7.12.2):
//  A (16x32 bf16): lane<16 -> M=lane, K = {0..7, 16..23}; lane>=16 -> K={8..15,24..31}
//  B (32x16 bf16): lane<16 -> N=lane, K = {0..7, 16..23}; lane>=16 -> N=lane-16, K={8..15,24..31}
//  C/D (16x16 f32): vgpr j -> M = j + 8*(lane>=16), N = lane&15
// ---------------------------------------------------------------------------
static __device__ __forceinline__ v8f wmma_bf16(bf16x16 a, bf16x16 b, v8f c) {
    return __builtin_amdgcn_wmma_f32_16x16x32_bf16(false, a, false, b, (short)0, c,
                                                   false, false);
}

static __device__ __forceinline__ v8f zero_v8f() {
    v8f z = {0.f, 0.f, 0.f, 0.f, 0.f, 0.f, 0.f, 0.f};
    return z;
}

// A fragment from a K-contiguous bf16 row (rowptr = row of matrix A for this lane's M)
static __device__ __forceinline__ bf16x16 load_afrag(const __bf16* rowptr, int k, int lane) {
    const int c0 = k + ((lane & 16) ? 8 : 0);
    bf16x8 lo = *(const bf16x8*)(rowptr + c0);
    bf16x8 hi = *(const bf16x8*)(rowptr + c0 + 16);
    bf16x16 a;
#pragma unroll
    for (int i = 0; i < 8; ++i) { a[i] = lo[i]; a[i + 8] = hi[i]; }
    return a;
}

// B fragment from B^T stored row-major (btrow = row n of B^T, K-contiguous)
static __device__ __forceinline__ bf16x16 load_bfrag(const __bf16* btrow, int k, int lane) {
    const int c0 = k + ((lane & 16) ? 8 : 0);
    bf16x8 lo = *(const bf16x8*)(btrow + c0);
    bf16x8 hi = *(const bf16x8*)(btrow + c0 + 16);
    bf16x16 b;
#pragma unroll
    for (int i = 0; i < 8; ++i) { b[i] = lo[i]; b[i + 8] = hi[i]; }
    return b;
}

// ---------------------------------------------------------------------------
// Async global->LDS staging (CDNA5 Tensor/async path, ISA ch.10.7 / 08_async_tensor):
//  per-lane GLOBAL_LOAD_ASYNC_TO_LDS_B128, tracked by ASYNCcnt — no VGPR round trip.
//  LDS offset = low 32 bits of the generic pointer (LDS aperture maps addr[31:0]).
// ---------------------------------------------------------------------------
static __device__ __forceinline__ void async_copy16(const void* gsrc, const void* ldst) {
    uint64_t ga = (uint64_t)(uintptr_t)gsrc;
    uint32_t la = (uint32_t)(uintptr_t)ldst;
    asm volatile("global_load_async_to_lds_b128 %0, %1, off"
                 :: "v"(la), "v"(ga)
                 : "memory");
}
static __device__ __forceinline__ void async_wait_all() {
    asm volatile("s_wait_asynccnt 0x0" ::: "memory");
}

// ---------------------------------------------------------------------------
// Kernel 0: small-tensor prep. Transposed bf16 weights + symmetrized filters.
//   wkmapT[kd][d]  = Wk_map[d][kd]
//   qe_b           = bf16(query_emb)
//   wqT[r][p][d]   = Wq[r][d][p],  wkT likewise
//   fbar[f][(n*3+m)*16+r] = (1/6) * sum_perm filters[f][P(n)][P(m)][r]
// ---------------------------------------------------------------------------
__global__ void prep_kernel(const float* __restrict__ filters,
                            const float* __restrict__ query_emb,
                            const float* __restrict__ Wk_map,
                            const float* __restrict__ Wq,
                            const float* __restrict__ Wk,
                            __bf16* __restrict__ wkmapT,
                            __bf16* __restrict__ qe_b,
                            __bf16* __restrict__ wqT,
                            __bf16* __restrict__ wkT,
                            float* __restrict__ fbar) {
    const int P6[6][3] = {{0,1,2},{0,2,1},{1,0,2},{1,2,0},{2,0,1},{2,1,0}};
    for (int i = blockIdx.x * blockDim.x + threadIdx.x; i < REL * PDIM * DD;
         i += gridDim.x * blockDim.x) {
        {   // wqT / wkT : i = ((r*64+p)*256 + d)
            int r = i >> 14, p = (i >> 8) & 63, d = i & 255;
            int src = (r * DD + d) * PDIM + p;
            wqT[i] = (__bf16)Wq[src];
            wkT[i] = (__bf16)Wk[src];
        }
        if (i < KDIM * DD) {           // wkmapT : i = kd*256 + d
            int kd = i >> 8, d = i & 255;
            wkmapT[i] = (__bf16)Wk_map[d * KDIM + kd];
        }
        if (i < QQ * KDIM) qe_b[i] = (__bf16)query_emb[i];
        if (i < NF * 144) {            // fbar
            int f = i / 144, j = i % 144;
            int nm = j >> 4, r = j & 15;
            int n = nm / 3, m = nm % 3;
            float s = 0.f;
#pragma unroll
            for (int p = 0; p < 6; ++p)
                s += filters[((f * GSZ + P6[p][n]) * GSZ + P6[p][m]) * REL + r];
            fbar[i] = s * (1.0f / 6.0f);
        }
    }
}

// ---------------------------------------------------------------------------
// Kernel 1: xT[b][d][n] = bf16(x[b][n][d])  (LDS-tiled 64x64 transpose)
// ---------------------------------------------------------------------------
__global__ void xt_kernel(const float* __restrict__ x, __bf16* __restrict__ xT) {
    const int bid = blockIdx.x;
    const int b = bid >> 6, rem = bid & 63;
    const int n0 = (rem >> 2) * 64, d0 = (rem & 3) * 64;
    __shared__ __bf16 tile[64][72];
#pragma unroll
    for (int i = 0; i < 16; ++i) {
        int idx = threadIdx.x + i * 256;
        int rr = idx >> 6, cc = idx & 63;
        tile[rr][cc] = (__bf16)x[((size_t)b * NN + n0 + rr) * DD + d0 + cc];
    }
    __syncthreads();
#pragma unroll
    for (int i = 0; i < 16; ++i) {
        int idx = threadIdx.x + i * 256;
        int rr = idx >> 6, cc = idx & 63;      // rr = d offset, cc = n offset
        xT[((size_t)b * DD + d0 + rr) * NN + n0 + cc] = tile[cc][rr];
    }
}

// ---------------------------------------------------------------------------
// Kernel 2: k = x @ Wk_map + sinusoidal PE  -> kb (bf16, [B*N][KD])
//   grid 2048 x 128 threads (4 waves). Wave w: rows blockIdx*64 + w*16, all 64 kd.
//   A is f32 x converted inline; B^T = wkmapT rows.
// ---------------------------------------------------------------------------
__global__ void kproj_kernel(const float* __restrict__ x,
                             const __bf16* __restrict__ wkmapT,
                             __bf16* __restrict__ kb) {
    const int lane = threadIdx.x & 31, wave = threadIdx.x >> 5;
    const int laneM = lane & 15, half = (lane & 16) ? 1 : 0;
    const int rowbase = blockIdx.x * 64 + wave * 16;
    const float* arow = x + (size_t)(rowbase + laneM) * DD;
    const int c0base = half ? 8 : 0;

    v8f acc[4];
#pragma unroll
    for (int t = 0; t < 4; ++t) acc[t] = zero_v8f();

    for (int k = 0; k < DD; k += 32) {
        const int c0 = k + c0base;
        float4 f0 = *(const float4*)(arow + c0);
        float4 f1 = *(const float4*)(arow + c0 + 4);
        float4 f2 = *(const float4*)(arow + c0 + 16);
        float4 f3 = *(const float4*)(arow + c0 + 20);
        bf16x16 a;
        a[0]=(__bf16)f0.x; a[1]=(__bf16)f0.y; a[2]=(__bf16)f0.z; a[3]=(__bf16)f0.w;
        a[4]=(__bf16)f1.x; a[5]=(__bf16)f1.y; a[6]=(__bf16)f1.z; a[7]=(__bf16)f1.w;
        a[8]=(__bf16)f2.x; a[9]=(__bf16)f2.y; a[10]=(__bf16)f2.z; a[11]=(__bf16)f2.w;
        a[12]=(__bf16)f3.x; a[13]=(__bf16)f3.y; a[14]=(__bf16)f3.z; a[15]=(__bf16)f3.w;
#pragma unroll
        for (int nt = 0; nt < 4; ++nt) {
            bf16x16 bb = load_bfrag(wkmapT + (nt * 16 + laneM) * DD, k, lane);
            acc[nt] = wmma_bf16(a, bb, acc[nt]);
        }
    }
    // add positional encoding, store bf16
#pragma unroll
    for (int nt = 0; nt < 4; ++nt) {
        const int kd = nt * 16 + laneM;
        const float inv = exp2f(-(float)(kd & ~1) * (13.287712379549449f / 64.0f));
#pragma unroll
        for (int j = 0; j < 8; ++j) {
            int row = rowbase + j + 8 * half;
            int tok = row & (NN - 1);
            float ang = (float)tok * inv;
            float pe = (kd & 1) ? __cosf(ang) : __sinf(ang);
            kb[(size_t)row * KDIM + kd] = (__bf16)(acc[nt][j] + pe);
        }
    }
}

// ---------------------------------------------------------------------------
// Kernel 3: scores = softmax_n( beta * qe @ k^T ), stored bf16 [B*Q][N]
//   grid 128*12, block 256 (8 waves). Block: one b, 16 query rows, all 1024 n.
//   Wave w covers n in [w*128, w*128+128) = 8 C tiles. Softmax via shfl + LDS.
// ---------------------------------------------------------------------------
__global__ void scores_kernel(const __bf16* __restrict__ qe_b,
                              const __bf16* __restrict__ kb,
                              __bf16* __restrict__ sc) {
    const int b = blockIdx.x / 12, qt = blockIdx.x % 12;
    const int tid = threadIdx.x, lane = tid & 31, wave = tid >> 5;
    const int laneM = lane & 15, half = (lane & 16) ? 1 : 0;
    const float beta = 0.125f;  // KD^-0.5

    const __bf16* arow = qe_b + (qt * 16 + laneM) * KDIM;
    v8f acc[8];
#pragma unroll
    for (int t = 0; t < 8; ++t) acc[t] = zero_v8f();

#pragma unroll
    for (int k = 0; k < KDIM; k += 32) {
        bf16x16 a = load_afrag(arow, k, lane);
#pragma unroll
        for (int t = 0; t < 8; ++t) {
            int n = wave * 128 + t * 16 + laneM;
            bf16x16 bb = load_bfrag(kb + ((size_t)b * NN + n) * KDIM, k, lane);
            acc[t] = wmma_bf16(a, bb, acc[t]);
        }
    }
    // scale by beta
#pragma unroll
    for (int t = 0; t < 8; ++t)
#pragma unroll
        for (int j = 0; j < 8; ++j) acc[t][j] *= beta;

    __shared__ float red[8][16];
    __shared__ float fin[16];

    // ---- row max ----
    float pmax[8];
#pragma unroll
    for (int j = 0; j < 8; ++j) {
        float m = -3.402823466e38f;
#pragma unroll
        for (int t = 0; t < 8; ++t) m = fmaxf(m, acc[t][j]);
#pragma unroll
        for (int mask = 1; mask < 16; mask <<= 1) m = fmaxf(m, __shfl_xor(m, mask));
        pmax[j] = m;
    }
    if (lane == 0)  { for (int j = 0; j < 8; ++j) red[wave][j]     = pmax[j]; }
    if (lane == 16) { for (int j = 0; j < 8; ++j) red[wave][8 + j] = pmax[j]; }
    __syncthreads();
    if (tid < 16) {
        float m = -3.402823466e38f;
        for (int w = 0; w < 8; ++w) m = fmaxf(m, red[w][tid]);
        fin[tid] = m;
    }
    __syncthreads();
    float rmax[8];
#pragma unroll
    for (int j = 0; j < 8; ++j) rmax[j] = fin[j + 8 * half];
    __syncthreads();

    // ---- exp + row sum ----
    float psum[8];
#pragma unroll
    for (int j = 0; j < 8; ++j) {
        float s = 0.f;
#pragma unroll
        for (int t = 0; t < 8; ++t) {
            acc[t][j] = __expf(acc[t][j] - rmax[j]);
            s += acc[t][j];
        }
#pragma unroll
        for (int mask = 1; mask < 16; mask <<= 1) s += __shfl_xor(s, mask);
        psum[j] = s;
    }
    if (lane == 0)  { for (int j = 0; j < 8; ++j) red[wave][j]     = psum[j]; }
    if (lane == 16) { for (int j = 0; j < 8; ++j) red[wave][8 + j] = psum[j]; }
    __syncthreads();
    if (tid < 16) {
        float s = 0.f;
        for (int w = 0; w < 8; ++w) s += red[w][tid];
        fin[tid] = s;
    }
    __syncthreads();
    float rinv[8];
#pragma unroll
    for (int j = 0; j < 8; ++j) rinv[j] = 1.0f / fin[j + 8 * half];

    // ---- store normalized scores (bf16) ----
#pragma unroll
    for (int t = 0; t < 8; ++t) {
        int n = wave * 128 + t * 16 + laneM;
#pragma unroll
        for (int j = 0; j < 8; ++j) {
            int q = qt * 16 + j + 8 * half;
            sc[((size_t)b * QQ + q) * NN + n] = (__bf16)(acc[t][j] * rinv[j]);
        }
    }
}

// ---------------------------------------------------------------------------
// Kernel 4: gobj = scores @ x  -> gq (bf16, [B*Q][D])
//   grid 128*12, block 128 (4 waves). Block: (b, 16 q rows); wave w: d in [w*64, w*64+64).
//   The 16 score rows (32 KB) are shared by all 4 waves -> stage once in LDS via
//   async global->LDS copies; B^T = xT rows streamed from global.
// ---------------------------------------------------------------------------
__global__ void value_kernel(const __bf16* __restrict__ sc,
                             const __bf16* __restrict__ xT,
                             __bf16* __restrict__ gq) {
    __shared__ __bf16 atile[16 * NN];   // 32 KB: scores[qt*16 .. +15][0..1023]
    const int b = blockIdx.x / 12, qt = blockIdx.x % 12;
    const int tid = threadIdx.x;
    const int lane = tid & 31, wave = tid >> 5;
    const int laneM = lane & 15, half = (lane & 16) ? 1 : 0;

    {   // async stage: 16 rows x 1024 bf16 contiguous = 2048 x 16B chunks
        const __bf16* src = sc + ((size_t)b * QQ + qt * 16) * NN;
#pragma unroll
        for (int i = 0; i < 16; ++i) {
            int e = (tid + i * 128) * 8;
            async_copy16(src + e, atile + e);
        }
        async_wait_all();
        __syncthreads();
    }

    v8f acc[4];
#pragma unroll
    for (int t = 0; t < 4; ++t) acc[t] = zero_v8f();

    const __bf16* arow = atile + laneM * NN;
    for (int k = 0; k < NN; k += 32) {
        bf16x16 a = load_afrag(arow, k, lane);
#pragma unroll
        for (int nt = 0; nt < 4; ++nt) {
            int d = wave * 64 + nt * 16 + laneM;
            bf16x16 bb = load_bfrag(xT + ((size_t)b * DD + d) * NN, k, lane);
            acc[nt] = wmma_bf16(a, bb, acc[nt]);
        }
    }
#pragma unroll
    for (int nt = 0; nt < 4; ++nt) {
        int d = wave * 64 + nt * 16 + laneM;
#pragma unroll
        for (int j = 0; j < 8; ++j) {
            int q = qt * 16 + j + 8 * half;
            gq[((size_t)b * QQ + q) * DD + d] = (__bf16)acc[nt][j];
        }
    }
}

// ---------------------------------------------------------------------------
// Kernel 5: relational core. Block = 16 (b,g) pairs = 48 gobj rows.
//   The 48x256 gobj tile (24 KB) is consumed identically by all waves for all
//   16 r's -> async-stage once in LDS (removes ~64x global read amplification).
//   For r in 0..15: Zq = gobj48 @ Wq[r], Zk = gobj48 @ Wk[r] (WMMA) -> LDS;
//   R[bg][n][m][r] = <zq_n, zk_m>; out[bg][f] = sum_{144} R * fbar[f].
// ---------------------------------------------------------------------------
__global__ void rel_kernel(const __bf16* __restrict__ gq,
                           const __bf16* __restrict__ wqT,
                           const __bf16* __restrict__ wkT,
                           const float* __restrict__ fbar,
                           float* __restrict__ out) {
    __shared__ __bf16 gt[48 * DD];      // 24 KB gobj tile
    __shared__ float zq[48][64];
    __shared__ float zk[48][64];
    __shared__ float R[16][144];        // [bg][(n*3+m)*16 + r]

    const int tid = threadIdx.x, lane = tid & 31, wave = tid >> 5;
    const int laneM = lane & 15, half = (lane & 16) ? 1 : 0;
    const size_t row0 = (size_t)blockIdx.x * 48;   // into gq [B*Q][D]

    {   // async stage: 48 x 256 bf16 contiguous = 1536 x 16B chunks
        const __bf16* src = gq + row0 * DD;
#pragma unroll
        for (int i = 0; i < 12; ++i) {
            int e = (tid + i * 128) * 8;
            async_copy16(src + e, gt + e);
        }
        async_wait_all();
        __syncthreads();
    }

    for (int r = 0; r < REL; ++r) {
        v8f aq[3], ak[3];
#pragma unroll
        for (int mt = 0; mt < 3; ++mt) { aq[mt] = zero_v8f(); ak[mt] = zero_v8f(); }

        const __bf16* bqRow = wqT + ((r * PDIM) + wave * 16 + laneM) * DD;
        const __bf16* bkRow = wkT + ((r * PDIM) + wave * 16 + laneM) * DD;

        for (int k = 0; k < DD; k += 32) {
            bf16x16 bq = load_bfrag(bqRow, k, lane);
            bf16x16 bk = load_bfrag(bkRow, k, lane);
#pragma unroll
            for (int mt = 0; mt < 3; ++mt) {
                bf16x16 a = load_afrag(gt + (mt * 16 + laneM) * DD, k, lane);
                aq[mt] = wmma_bf16(a, bq, aq[mt]);
                ak[mt] = wmma_bf16(a, bk, ak[mt]);
            }
        }
        // spill 48x64 Zq/Zk tiles to LDS
#pragma unroll
        for (int mt = 0; mt < 3; ++mt) {
            int p = wave * 16 + laneM;
#pragma unroll
            for (int j = 0; j < 8; ++j) {
                int M = mt * 16 + j + 8 * half;
                zq[M][p] = aq[mt][j];
                zk[M][p] = ak[mt][j];
            }
        }
        __syncthreads();
        for (int idx = tid; idx < 144; idx += 128) {
            int bg = idx / 9, nm = idx % 9;
            int n = nm / 3, m = nm % 3;
            const float* zqr = zq[bg * 3 + n];
            const float* zkr = zk[bg * 3 + m];
            float s = 0.f;
#pragma unroll
            for (int p = 0; p < PDIM; ++p) s += zqr[p] * zkr[p];
            R[bg][nm * 16 + r] = s;
        }
        __syncthreads();
    }

    // final contraction: each thread owns one f (fbar row stays in registers'
    // reach / L0) and 8 bg accumulators.
    {
        const int f = tid & 63;
        const int bg0 = tid >> 6;            // 0 or 1
        const float* fb = fbar + f * 144;
        float accо[8];
#pragma unroll
        for (int s = 0; s < 8; ++s) accо[s] = 0.f;
        for (int j = 0; j < 144; ++j) {
            float fv = fb[j];
#pragma unroll
            for (int s = 0; s < 8; ++s) accо[s] += R[bg0 + 2 * s][j] * fv;
        }
#pragma unroll
        for (int s = 0; s < 8; ++s) {
            int bg = bg0 + 2 * s;
            out[((size_t)blockIdx.x * 16 + bg) * NF + f] = accо[s];
        }
    }
}

// ---------------------------------------------------------------------------
// Host launcher
// ---------------------------------------------------------------------------
extern "C" void kernel_launch(void* const* d_in, const int* in_sizes, int n_in,
                              void* d_out, int out_size, void* d_ws, size_t ws_size,
                              hipStream_t stream) {
    const float* x         = (const float*)d_in[0];
    const float* filters   = (const float*)d_in[1];
    const float* query_emb = (const float*)d_in[2];
    const float* Wk_map    = (const float*)d_in[3];
    const float* Wq        = (const float*)d_in[4];
    const float* Wk        = (const float*)d_in[5];
    float* out = (float*)d_out;

    // workspace layout (bytes, 32-aligned)
    char* ws = (char*)d_ws;
    const size_t off_xT     = 0;                                 // 64 MB bf16
    const size_t off_kb     = off_xT     + (size_t)BB*DD*NN*2;   // 16 MB bf16
    const size_t off_sc     = off_kb     + (size_t)BB*NN*KDIM*2; // 48 MB bf16
    const size_t off_gq     = off_sc     + (size_t)BB*QQ*NN*2;   // 12 MB bf16
    const size_t off_wkmapT = off_gq     + (size_t)BB*QQ*DD*2;
    const size_t off_qe     = off_wkmapT + (size_t)KDIM*DD*2 + 224;   // only 32B-aligned
    const size_t off_wqT    = off_qe     + (size_t)QQ*KDIM*2 + 128;
    const size_t off_wkT    = off_wqT    + (size_t)REL*PDIM*DD*2;
    const size_t off_fbar   = off_wkT    + (size_t)REL*PDIM*DD*2;
    const size_t need       = off_fbar   + (size_t)NF*144*4;
    if (ws_size < need) return;  // workspace assumption: ~148 MB

    __bf16* xT     = (__bf16*)(ws + off_xT);
    __bf16* kb     = (__bf16*)(ws + off_kb);
    __bf16* sc     = (__bf16*)(ws + off_sc);
    __bf16* gq     = (__bf16*)(ws + off_gq);
    __bf16* wkmapT = (__bf16*)(ws + off_wkmapT);
    __bf16* qe_b   = (__bf16*)(ws + off_qe);
    __bf16* wqT    = (__bf16*)(ws + off_wqT);
    __bf16* wkT    = (__bf16*)(ws + off_wkT);
    float*  fbar   = (float*)(ws + off_fbar);

    prep_kernel<<<512, 256, 0, stream>>>(filters, query_emb, Wk_map, Wq, Wk,
                                         wkmapT, qe_b, wqT, wkT, fbar);
    xt_kernel<<<BB * 16 * 4, 256, 0, stream>>>(x, xT);
    kproj_kernel<<<(BB * NN) / 64, 128, 0, stream>>>(x, wkmapT, kb);
    scores_kernel<<<BB * (QQ / 16), 256, 0, stream>>>(qe_b, kb, sc);
    value_kernel<<<BB * (QQ / 16), 128, 0, stream>>>(sc, xT, gq);
    rel_kernel<<<(BB * GG) / 16, 128, 0, stream>>>(gq, wqT, wkT, fbar, out);
}
